// VisionTransformerDetFastForBeit_50551765074260
// MI455X (gfx1250) — matrix operand order repaired and checked by reading, not compiled, not run on hardware
//
#include <hip/hip_runtime.h>
#include <hip/hip_bf16.h>

// ---------------------------------------------------------------------------
// BEiT window attention block for MI455X (gfx1250, wave32, WMMA).
// All matmuls use v_wmma_f32_16x16x32_bf16 (f32 accumulate).
// ---------------------------------------------------------------------------

typedef __attribute__((ext_vector_type(16))) __bf16 v16bf;
typedef __attribute__((ext_vector_type(8)))  float  v8f;

union AFrag { v16bf v; unsigned u[8]; };

__device__ __forceinline__ unsigned short f2bf(float x) {
  union { float f; unsigned u; } v; v.f = x;
  unsigned r = v.u + 0x7FFFu + ((v.u >> 16) & 1u);   // round-to-nearest-even
  return (unsigned short)(r >> 16);
}
__device__ __forceinline__ unsigned pack_bf16(float a, float b) {
  return (unsigned)f2bf(a) | ((unsigned)f2bf(b) << 16);
}
__device__ __forceinline__ v8f wmma_bf16(const AFrag& a, const AFrag& b, v8f c) {
  return __builtin_amdgcn_wmma_f32_16x16x32_bf16(false, a.v, false, b.v,
                                                 (short)0, c, false, false);
}

// ---------------------------------------------------------------------------
// Small prep kernels
// ---------------------------------------------------------------------------
__global__ void cvt_bf16_kernel(const float* __restrict__ src,
                                unsigned short* __restrict__ dst, int n) {
  int i = blockIdx.x * 256 + threadIdx.x;
  if (i < n) dst[i] = f2bf(src[i]);
}

__global__ void build_qkv_bias_kernel(const float* __restrict__ qb,
                                      const float* __restrict__ vb,
                                      float* __restrict__ out) {
  int i = blockIdx.x * 256 + threadIdx.x;
  if (i < 768) { out[i] = qb[i]; out[768 + i] = 0.f; out[1536 + i] = vb[i]; }
}

__global__ void build_rpb_kernel(const float* __restrict__ tab,
                                 const int* __restrict__ idx,
                                 float* __restrict__ out) {
  int i = blockIdx.x * 256 + threadIdx.x;      // over 196*196
  if (i < 196 * 196) {
    int id = idx[i];
    #pragma unroll
    for (int h = 0; h < 12; h++) out[(size_t)h * 38416 + i] = tab[id * 12 + h];
  }
}

// ---------------------------------------------------------------------------
// Tiled bf16 WMMA GEMM:  C[M][N] = A[M][K] (fp32) * W[N][K]^T (bf16) + bias[N]
// 128x128 block tile, K-step 32, double-buffered LDS, 8 waves (4x2).
// ---------------------------------------------------------------------------
#define GTM 128
#define GTN 128
#define GTK 32

__global__ __launch_bounds__(256) void gemm_bf16_wmma(
    const float* __restrict__ A, const unsigned short* __restrict__ Wb,
    const float* __restrict__ bias, float* __restrict__ C,
    int M, int N, int K) {
  __shared__ unsigned sA[2][GTM][GTK / 2];   // K-pairs along rows  (16 KB)
  __shared__ unsigned sB[2][GTK / 2][GTN];   // [kpair][n]          (16 KB)

  const int tid = threadIdx.x;
  const int lane = tid & 31, wid = tid >> 5;
  const int m0 = blockIdx.x * GTM, n0 = blockIdx.y * GTN;
  const int wm = (wid & 3) * 32;             // 4 waves along M
  const int wn = (wid >> 2) * 64;            // 2 waves along N
  const int h4 = (lane < 16) ? 0 : 4;
  const int h8 = (lane < 16) ? 0 : 8;
  const unsigned* W32 = (const unsigned*)Wb; // bf16 pairs along K

  v8f acc[2][4];
  #pragma unroll
  for (int i = 0; i < 2; i++)
    #pragma unroll
    for (int j = 0; j < 4; j++) acc[i][j] = {};

  const int nsteps = K / GTK;

  auto loadStep = [&](int buf, int ks) {
    const int k0 = ks * GTK;
    // A slab: 128 rows x 16 k-pairs (convert fp32 -> packed bf16)
    for (int i = tid; i < GTM * 16; i += 256) {
      int r = i >> 4, kp = i & 15;
      int gm = m0 + r;
      float2 f = make_float2(0.f, 0.f);
      if (gm < M) f = *(const float2*)(A + (size_t)gm * K + k0 + 2 * kp);
      sA[buf][r][kp] = pack_bf16(f.x, f.y);
    }
    // B slab: 16 k-pairs x 128 n  (weights already bf16, pairs contiguous)
    for (int i = tid; i < GTN * 16; i += 256) {
      int n = i >> 4, kp = i & 15;
      sB[buf][kp][n] = W32[((size_t)(n0 + n) * K + k0) / 2 + kp];
    }
  };

  loadStep(0, 0);
  __syncthreads();

  for (int ks = 0; ks < nsteps; ks++) {
    const int buf = ks & 1;
    if (ks + 1 < nsteps) loadStep(buf ^ 1, ks + 1);

    AFrag a[2], b[4];
    #pragma unroll
    for (int mt = 0; mt < 2; mt++) {
      int row = wm + mt * 16 + (lane & 15);
      #pragma unroll
      for (int v = 0; v < 8; v++) {
        int p = (v & 3) + ((v >> 2) << 3) + h4;   // A K-pair index
        a[mt].u[v] = sA[buf][row][p];
      }
    }
    #pragma unroll
    for (int nt = 0; nt < 4; nt++) {
      int col = wn + nt * 16 + (lane & 15);
      #pragma unroll
      for (int v = 0; v < 8; v++) b[nt].u[v] = sB[buf][(v & 3) + ((v >> 2) << 3) + h4][col];
    }
    #pragma unroll
    for (int mt = 0; mt < 2; mt++)
      #pragma unroll
      for (int nt = 0; nt < 4; nt++)
        acc[mt][nt] = wmma_bf16(a[mt], b[nt], acc[mt][nt]);
    __syncthreads();
  }

  // Epilogue (C/D layout: VGPR r -> M = r + 8*(lane>=16), N = lane&15)
  #pragma unroll
  for (int mt = 0; mt < 2; mt++)
    #pragma unroll
    for (int nt = 0; nt < 4; nt++) {
      int col = n0 + wn + nt * 16 + (lane & 15);
      float bv = bias[col];
      #pragma unroll
      for (int r = 0; r < 8; r++) {
        int row = m0 + wm + mt * 16 + r + ((lane < 16) ? 0 : 8);
        if (row < M) C[(size_t)row * N + col] = acc[mt][nt][r] + bv;
      }
    }
}

// ---------------------------------------------------------------------------
// Fused attention: one workgroup per (b, head); 13 waves, one 16-row q-tile
// per wave; scores kept in registers; softmax via 16-lane shfl reductions.
// ---------------------------------------------------------------------------
#define HWA   196
#define NKT   13
#define PCOLS 224   // padded key dim (7 * 32)

__global__ __launch_bounds__(416) void attn_kernel(
    const float* __restrict__ qkv,       // [B*196][2304]
    const float* __restrict__ relh_tab,  // [27][64]
    const float* __restrict__ relw_tab,  // [27][64]
    const float* __restrict__ rpb,       // [12][196][196]
    float* __restrict__ outp) {          // [B*196][768]
  extern __shared__ unsigned char smem[];
  unsigned*       sQp  = (unsigned*)smem;                       // [208][32]  q pairs
  unsigned*       sKp  = sQp + 208 * 32;                        // [32][224]  K^T pairs
  unsigned*       sVp  = sKp + 32 * 224;                        // [64][112]  V^T key-pairs
  float*          sRel = (float*)(sVp + 64 * 112);              // [208][32]  relh|relw
  unsigned short* sP   = (unsigned short*)(sRel + 208 * 32);    // [13][16][224]

  const int b = blockIdx.x / 12, head = blockIdx.x % 12;
  const int tid = threadIdx.x, lane = tid & 31, wv = tid >> 5;
  const size_t QROW = 2304;
  const float* qbase = qkv + (size_t)b * HWA * QROW + head * 64;

  // ---- fill Q (pair-packed bf16), zero-padded rows ----
  for (int i = tid; i < 208 * 32; i += 416) {
    int r = i >> 5, p = i & 31;
    unsigned u = 0;
    if (r < HWA) {
      const float* q = qbase + (size_t)r * QROW + 2 * p;
      u = pack_bf16(q[0], q[1]);
    }
    sQp[i] = u;
  }
  // ---- fill K^T: sKp[hd_pair][key] ----
  for (int i = tid; i < 32 * 224; i += 416) {
    int hp = i / 224, key = i % 224;
    unsigned u = 0;
    if (key < HWA) {
      const float* k = qbase + 768 + (size_t)key * QROW + 2 * hp;
      u = pack_bf16(k[0], k[1]);
    }
    sKp[i] = u;
  }
  // ---- fill V^T: sVp[hd][key_pair] ----
  for (int i = tid; i < 64 * 112; i += 416) {
    int hd = i / 112, kp = i % 112;
    int k0 = 2 * kp, k1 = 2 * kp + 1;
    float v0 = (k0 < HWA) ? qbase[1536 + (size_t)k0 * QROW + hd] : 0.f;
    float v1 = (k1 < HWA) ? qbase[1536 + (size_t)k1 * QROW + hd] : 0.f;
    sVp[i] = pack_bf16(v0, v1);
  }
  // ---- rel_h / rel_w dot products (fp32): sRel[row][0..13]=relh, [16..29]=relw
  for (int i = tid; i < 208 * 32; i += 416) {
    int r = i >> 5, c = i & 31;
    float acc = 0.f;
    if (r < HWA) {
      int qh = r / 14, qw = r % 14;
      const float* q = qbase + (size_t)r * QROW;
      if (c < 14) {
        const float* rh = relh_tab + (size_t)(qh - c + 13) * 64;
        #pragma unroll 8
        for (int j = 0; j < 64; j++) acc += q[j] * rh[j];
      } else if (c >= 16 && c < 30) {
        const float* rw = relw_tab + (size_t)(qw - (c - 16) + 13) * 64;
        #pragma unroll 8
        for (int j = 0; j < 64; j++) acc += q[j] * rw[j];
      }
    }
    sRel[i] = acc;
  }
  __syncthreads();

  // ---- stage 1: S = Q K^T * scale + relh + relw + rpb ----
  const int qt = wv;                       // this wave's query tile (0..12)
  const int mrow = lane & 15;
  const int h4 = (lane < 16) ? 0 : 4;
  const int h8 = (lane < 16) ? 0 : 8;

  AFrag aq[2];
  #pragma unroll
  for (int ks = 0; ks < 2; ks++)
    #pragma unroll
    for (int v = 0; v < 8; v++) {
      int p = (v & 3) + ((v >> 2) << 3) + h4 + ks * 16;
      aq[ks].u[v] = sQp[(qt * 16 + mrow) * 32 + p];
    }

  float S[NKT][8];
  const float scale = 0.125f;              // 64^-0.5
  for (int kt = 0; kt < NKT; kt++) {
    AFrag bk[2];
    #pragma unroll
    for (int ks = 0; ks < 2; ks++)
      #pragma unroll
      for (int v = 0; v < 8; v++)
        bk[ks].u[v] = sKp[(ks * 16 + (v & 3) + ((v >> 2) << 3) + h4) * 224 + kt * 16 + (lane & 15)];
    v8f c = {};
    c = wmma_bf16(aq[0], bk[0], c);
    c = wmma_bf16(aq[1], bk[1], c);

    int key = kt * 16 + (lane & 15);
    int kh = key / 14, kw = key % 14;
    #pragma unroll
    for (int r = 0; r < 8; r++) {
      int qrow = qt * 16 + r + ((lane < 16) ? 0 : 8);
      float s = -1e30f;
      if (key < HWA && qrow < HWA)
        s = c[r] * scale + sRel[qrow * 32 + kh] + sRel[qrow * 32 + 16 + kw]
            + rpb[((size_t)head * HWA + qrow) * HWA + key];
      S[kt][r] = s;
    }
  }

  // ---- softmax over 196 keys (13 register tiles x 16-lane groups) ----
  float inv[8];
  #pragma unroll
  for (int r = 0; r < 8; r++) {
    float m = -1e30f;
    #pragma unroll
    for (int kt = 0; kt < NKT; kt++) m = fmaxf(m, S[kt][r]);
    for (int d = 1; d < 16; d <<= 1) m = fmaxf(m, __shfl_xor(m, d, 32));
    float sum = 0.f;
    #pragma unroll
    for (int kt = 0; kt < NKT; kt++) {
      float p = __expf(S[kt][r] - m);
      S[kt][r] = p;
      sum += p;
    }
    for (int d = 1; d < 16; d <<= 1) sum += __shfl_xor(sum, d, 32);
    inv[r] = 1.f / sum;
  }

  // ---- write P (bf16) to this wave's LDS slab, zero pad cols 208..223 ----
  unsigned short* myP = sP + wv * 16 * PCOLS;
  #pragma unroll
  for (int r = 0; r < 8; r++) {
    int m = r + ((lane < 16) ? 0 : 8);
    #pragma unroll
    for (int kt = 0; kt < NKT; kt++)
      myP[m * PCOLS + kt * 16 + (lane & 15)] = f2bf(S[kt][r]);
  }
  for (int i = lane; i < 16 * 16; i += 32)
    myP[(i >> 4) * PCOLS + 208 + (i & 15)] = 0;
  __syncthreads();

  // ---- stage 2: O = P V  (7 K-steps of 32, 4 output tiles of 16 cols) ----
  v8f o[4];
  #pragma unroll
  for (int nt = 0; nt < 4; nt++) o[nt] = {};
  const unsigned* myP32 = (const unsigned*)myP;
  for (int kc = 0; kc < 7; kc++) {
    AFrag ap;
    #pragma unroll
    for (int v = 0; v < 8; v++) {
      int p = (v & 3) + ((v >> 2) << 3) + h4 + kc * 16;
      ap.u[v] = myP32[mrow * (PCOLS / 2) + p];
    }
    #pragma unroll
    for (int nt = 0; nt < 4; nt++) {
      AFrag bv;
      #pragma unroll
      for (int v = 0; v < 8; v++)
        bv.u[v] = sVp[(nt * 16 + (lane & 15)) * 112 + kc * 16 + (v & 3) + ((v >> 2) << 3) + h4];
      o[nt] = wmma_bf16(ap, bv, o[nt]);
    }
  }

  // ---- scale by 1/sum and store ----
  #pragma unroll
  for (int nt = 0; nt < 4; nt++) {
    int col = head * 64 + nt * 16 + (lane & 15);
    #pragma unroll
    for (int r = 0; r < 8; r++) {
      int qrow = qt * 16 + r + ((lane < 16) ? 0 : 8);
      if (qrow < HWA)
        outp[((size_t)b * HWA + qrow) * 768 + col] = o[nt][r] * inv[r];
    }
  }
}

// ---------------------------------------------------------------------------
// Launch
// ---------------------------------------------------------------------------
extern "C" void kernel_launch(void* const* d_in, const int* in_sizes, int n_in,
                              void* d_out, int out_size, void* d_ws, size_t ws_size,
                              hipStream_t stream) {
  const float* x         = (const float*)d_in[0];   // [200,14,14,768]
  const float* qkv_w     = (const float*)d_in[1];   // [2304,768]
  const float* q_bias    = (const float*)d_in[2];   // [768]
  const float* v_bias    = (const float*)d_in[3];   // [768]
  const float* rel_pos_h = (const float*)d_in[4];   // [27,64]
  const float* rel_pos_w = (const float*)d_in[5];   // [27,64]
  const float* rpb_table = (const float*)d_in[6];   // [732,12]
  const float* proj_w    = (const float*)d_in[7];   // [768,768]
  const float* proj_b    = (const float*)d_in[8];   // [768]
  const int*   rp_idx    = (const int*)d_in[9];     // [196,196]

  char* ws = (char*)d_ws;
  size_t off = 0;
  float* qkv = (float*)(ws + off);            off += (size_t)39200 * 2304 * 4;
  float* attn_out = (float*)(ws + off);       off += (size_t)39200 * 768 * 4;
  unsigned short* qkv_wb = (unsigned short*)(ws + off);  off += (size_t)2304 * 768 * 2;
  unsigned short* proj_wb = (unsigned short*)(ws + off); off += (size_t)768 * 768 * 2;
  float* qkv_bias = (float*)(ws + off);       off += 2304 * 4;
  float* rpb_full = (float*)(ws + off);       off += (size_t)12 * 196 * 196 * 4;

  cvt_bf16_kernel<<<(2304 * 768 + 255) / 256, 256, 0, stream>>>(qkv_w, qkv_wb, 2304 * 768);
  cvt_bf16_kernel<<<(768 * 768 + 255) / 256, 256, 0, stream>>>(proj_w, proj_wb, 768 * 768);
  build_qkv_bias_kernel<<<3, 256, 0, stream>>>(q_bias, v_bias, qkv_bias);
  build_rpb_kernel<<<(196 * 196 + 255) / 256, 256, 0, stream>>>(rpb_table, rp_idx, rpb_full);

  // QKV projection: [39200,768] x [2304,768]^T
  dim3 g1((39200 + GTM - 1) / GTM, 2304 / GTN);
  gemm_bf16_wmma<<<g1, 256, 0, stream>>>(x, qkv_wb, qkv_bias, qkv, 39200, 2304, 768);

  // Attention: 200 batches x 12 heads, 13 waves, ~204 KB dynamic LDS
  size_t smem = (size_t)(208 * 32 + 32 * 224 + 64 * 112 + 208 * 32) * 4
              + (size_t)13 * 16 * PCOLS * 2;
  hipFuncSetAttribute((const void*)attn_kernel,
                      hipFuncAttributeMaxDynamicSharedMemorySize, (int)smem);
  attn_kernel<<<200 * 12, 416, smem, stream>>>(qkv, rel_pos_h, rel_pos_w, rpb_full, attn_out);

  // Output projection: [39200,768] x [768,768]^T
  dim3 g2((39200 + GTM - 1) / GTM, 768 / GTN);
  gemm_bf16_wmma<<<g2, 256, 0, stream>>>(attn_out, proj_wb, proj_b, (float*)d_out,
                                         39200, 768, 768);
}
